// SigLipLossWithAmbiguity_29635274342858
// MI455X (gfx1250) — matrix operand order repaired and checked by reading, not compiled, not run on hardware
//
#include <hip/hip_runtime.h>
#include <hip/hip_bf16.h>
#include <math.h>
#include <stdint.h>

typedef __attribute__((ext_vector_type(16))) _Float16 v16h;
typedef __attribute__((ext_vector_type(8)))  float    v8f;
typedef __attribute__((ext_vector_type(4)))  unsigned int v4u;

#define S_IMG 16384
#define N_TXT 4096
#define DIM   512
#define SA    40   // LDS tile row stride in f16 (32 data + 8 pad -> 80B, bank-conflict free)

// ---------------------------------------------------------------------------
// One wave-wide async global->LDS 16B/lane copy (CDNA5 async path, ASYNCcnt).
// vdst = per-lane LDS byte address, vaddr = per-lane 64-bit global address.
// ---------------------------------------------------------------------------
__device__ __forceinline__ void async_b128(const _Float16* gaddr, unsigned lds_off) {
    asm volatile("global_load_async_to_lds_b128 %0, %1, off"
                 :: "v"(lds_off), "v"((unsigned long long)(uintptr_t)gaddr)
                 : "memory");
}

// ---------------------------------------------------------------------------
// Row L2-normalize: f32 row to d_out, f16 row to workspace (for WMMA GEMMs).
// One 128-thread block per row (4 elements/thread).
// ---------------------------------------------------------------------------
__global__ __launch_bounds__(128) void k_normalize(const float* __restrict__ in,
                                                   float* __restrict__ outF,
                                                   _Float16* __restrict__ outH) {
    const int row = blockIdx.x;
    const float* p = in + (size_t)row * DIM;
    float vals[4];
    float ss = 0.f;
#pragma unroll
    for (int j = 0; j < 4; ++j) {
        float v = p[threadIdx.x + j * 128];
        vals[j] = v;
        ss += v * v;
    }
#pragma unroll
    for (int off = 16; off > 0; off >>= 1) ss += __shfl_xor(ss, off, 32);
    __shared__ float red[4];
    const int lane = threadIdx.x & 31, wv = threadIdx.x >> 5;
    if (lane == 0) red[wv] = ss;
    __syncthreads();
    const float tot = red[0] + red[1] + red[2] + red[3];
    const float inv = 1.0f / (sqrtf(tot) + 1e-12f);
    float*    of = outF + (size_t)row * DIM;
    _Float16* oh = outH + (size_t)row * DIM;
#pragma unroll
    for (int j = 0; j < 4; ++j) {
        float z = vals[j] * inv;
        of[threadIdx.x + j * 128] = z;
        oh[threadIdx.x + j * 128] = (_Float16)z;
    }
}

// ---------------------------------------------------------------------------
// WMMA GEMM with double-buffered async-LDS staging:
//   C[M,Ncols] = (A[M,DIM] * B[Ncols,DIM]^T) * t + bias   (f16 in, f32 out)
// 256 threads = 8 waves; block tile 128x128; wave tile 64x32 = 4x2 WMMA tiles.
// Per K-step (32), A/B 128x32 tiles staged via global_load_async_to_lds_b128;
// stage s+1 is issued before waiting (s_wait_asynccnt 4) on stage s.
// FUSE_LOSS additionally accumulates sum(softplus(-(label*logit))),
// label = +1 on the diagonal, -1 elsewhere.
// ---------------------------------------------------------------------------
template <bool FUSE_LOSS>
__global__ __launch_bounds__(256) void k_gemm(const _Float16* __restrict__ A,
                                              const _Float16* __restrict__ B,
                                              float* __restrict__ C,
                                              int Ncols,
                                              const float* __restrict__ logt_p,
                                              const float* __restrict__ bias_p,
                                              float* __restrict__ lossAcc) {
    __shared__ __align__(16) _Float16 Ash[2][128 * SA];
    __shared__ __align__(16) _Float16 Bsh[2][128 * SA];

    const int tid   = threadIdx.x;
    const int wave  = tid >> 5;
    const int lane  = tid & 31;
    const int waveM = wave & 1;   // 2 waves along M
    const int waveN = wave >> 1;  // 4 waves along N
    const int mBase = blockIdx.y * 128;
    const int nBase = blockIdx.x * 128;
    const int r  = lane & 15;
    const int hi = lane >> 4;

    // async staging: each thread owns two 16B chunks of each tile
    const int c0 = tid, c1 = tid + 256;
    const int row0 = c0 >> 2, kc0 = (c0 & 3) * 8;   // f16 units within the 32-wide K slab
    const int row1 = c1 >> 2, kc1 = (c1 & 3) * 8;

    const _Float16* Ag = A + (size_t)mBase * DIM;
    const _Float16* Bg = B + (size_t)nBase * DIM;

    auto issue_stage = [&](int buf, int k0) {
        async_b128(Ag + (size_t)row0 * DIM + k0 + kc0,
                   (unsigned)(uintptr_t)&Ash[buf][row0 * SA + kc0]);
        async_b128(Ag + (size_t)row1 * DIM + k0 + kc1,
                   (unsigned)(uintptr_t)&Ash[buf][row1 * SA + kc1]);
        async_b128(Bg + (size_t)row0 * DIM + k0 + kc0,
                   (unsigned)(uintptr_t)&Bsh[buf][row0 * SA + kc0]);
        async_b128(Bg + (size_t)row1 * DIM + k0 + kc1,
                   (unsigned)(uintptr_t)&Bsh[buf][row1 * SA + kc1]);
    };

    v8f acc[4][2] = {};

    issue_stage(0, 0);

    for (int s = 0; s < DIM / 32; ++s) {
        if (s + 1 < DIM / 32) {
            issue_stage((s + 1) & 1, (s + 1) * 32);
            // previous stage (4 ops) complete; next stage (4 ops) in flight
            asm volatile("s_wait_asynccnt 0x4" ::: "memory");
        } else {
            asm volatile("s_wait_asynccnt 0x0" ::: "memory");
        }
        __syncthreads();   // stage s visible to all waves

        const _Float16* As = Ash[s & 1];
        const _Float16* Bs = Bsh[s & 1];
        v16h a[4], b[2];
        // A fragment (ISA 16-bit A 16x32): lanes<16 hold K {0..7, 16..23}
#pragma unroll
        for (int mt = 0; mt < 4; ++mt) {
            const _Float16* p = As + (waveM * 64 + mt * 16 + r) * SA + hi * 8;
            v4u* d = (v4u*)&a[mt];
            d[0] = *(const v4u*)p;
            d[1] = *(const v4u*)(p + 16);
        }
        // B fragment (same as A): lanes<16 hold K {0..7, 16..23}
#pragma unroll
        for (int nt = 0; nt < 2; ++nt) {
            const _Float16* p = Bs + (waveN * 32 + nt * 16 + r) * SA + hi * 8;
            v4u* d = (v4u*)&b[nt];
            d[0] = *(const v4u*)p;
            d[1] = *(const v4u*)(p + 16);
        }
#pragma unroll
        for (int mt = 0; mt < 4; ++mt)
#pragma unroll
            for (int nt = 0; nt < 2; ++nt)
                acc[mt][nt] = __builtin_amdgcn_wmma_f32_16x16x32_f16(
                    false, a[mt], false, b[nt], (short)0, acc[mt][nt], false, false);

        __syncthreads();   // all reads of buf (s&1) done before stage s+2 overwrites it
    }

    const float t    = expf(*logt_p);
    const float bias = *bias_p;
    float part = 0.f;

    // D layout: lane (mod 16) = column, VGPR v = row (+8 for lanes 16..31)
#pragma unroll
    for (int mt = 0; mt < 4; ++mt) {
#pragma unroll
        for (int nt = 0; nt < 2; ++nt) {
            const int rowD = mBase + waveM * 64 + mt * 16 + hi * 8;
            const int col  = nBase + waveN * 32 + nt * 16 + r;
#pragma unroll
            for (int v = 0; v < 8; ++v) {
                const float val = acc[mt][nt][v] * t + bias;
                C[(size_t)(rowD + v) * Ncols + col] = val;
                if (FUSE_LOSS) {
                    const float lab = ((rowD + v) == col) ? 1.f : -1.f;
                    const float z = -lab * val;
                    part += fmaxf(z, 0.f) + log1pf(expf(-fabsf(z)));
                }
            }
        }
    }

    if (FUSE_LOSS) {
#pragma unroll
        for (int off = 16; off > 0; off >>= 1) part += __shfl_xor(part, off, 32);
        __shared__ float redl[8];
        if (lane == 0) redl[wave] = part;
        __syncthreads();
        if (tid == 0) {
            float sum = 0.f;
#pragma unroll
            for (int w = 0; w < 8; ++w) sum += redl[w];
            atomicAdd(lossAcc, sum);
        }
    }
}

// ---------------------------------------------------------------------------
// Per-call re-init: argmin packs to all-ones, loss accumulator to zero.
// ---------------------------------------------------------------------------
__global__ void k_init(unsigned long long* __restrict__ packed, float* __restrict__ lossAcc) {
    const int i = blockIdx.x * blockDim.x + threadIdx.x;
    if (i < N_TXT) packed[i] = ~0ull;
    if (i == 0) *lossAcc = 0.f;
}

// ---------------------------------------------------------------------------
// True-positive losses + grouped argmin via (loss_bits << 32 | index) u64 min.
// softplus(.) > 0 so positive-f32 bit patterns are order-preserving.
// ---------------------------------------------------------------------------
__global__ void k_tp(const float* __restrict__ allp, const int* __restrict__ key,
                     unsigned long long* __restrict__ packed) {
    const int i = blockIdx.x * blockDim.x + threadIdx.x;
    if (i >= S_IMG) return;
    const int k = key[i];
    const float logit = allp[(size_t)i * N_TXT + k];
    const float z = -logit;
    const float loss = fmaxf(z, 0.f) + log1pf(expf(-fabsf(z)));
    const unsigned long long pk =
        ((unsigned long long)__float_as_uint(loss) << 32) | (unsigned int)i;
    atomicMin(&packed[k], pk);
}

// ---------------------------------------------------------------------------
// Decode winners: write selected index (as float) and gather/zero the f16 row.
// 128 threads * 8B = one 512-f16 row.
// ---------------------------------------------------------------------------
__global__ __launch_bounds__(128) void k_select(const unsigned long long* __restrict__ packed,
                                                const _Float16* __restrict__ zimgH,
                                                _Float16* __restrict__ zselH,
                                                float* __restrict__ selOut) {
    const int n = blockIdx.x;
    const unsigned long long p = packed[n];
    const int sel = (p == ~0ull) ? -1 : (int)(unsigned int)(p & 0xffffffffull);
    if (threadIdx.x == 0) selOut[n] = (float)sel;
    unsigned long long v = 0ull;
    if (sel >= 0)
        v = ((const unsigned long long*)(zimgH + (size_t)sel * DIM))[threadIdx.x];
    ((unsigned long long*)(zselH + (size_t)n * DIM))[threadIdx.x] = v;
}

__global__ void k_finalize(const float* __restrict__ lossAcc, float* __restrict__ out) {
    if (blockIdx.x == 0 && threadIdx.x == 0) out[0] = lossAcc[0] / (float)N_TXT;
}

// ---------------------------------------------------------------------------
extern "C" void kernel_launch(void* const* d_in, const int* in_sizes, int n_in,
                              void* d_out, int out_size, void* d_ws, size_t ws_size,
                              hipStream_t stream) {
    (void)in_sizes; (void)n_in; (void)out_size; (void)ws_size;
    const float* img  = (const float*)d_in[0];
    const float* txt  = (const float*)d_in[1];
    const int*   key  = (const int*)d_in[2];
    const float* logt = (const float*)d_in[3];
    const float* bias = (const float*)d_in[4];

    float* out      = (float*)d_out;
    float* outLoss  = out;                                   // [1]
    float* outSel   = out + 1;                               // [N]
    float* outZimg  = out + 1 + N_TXT;                       // [S*DIM]
    float* outZtxt  = outZimg + (size_t)S_IMG * DIM;         // [N*DIM]
    float* outAllp  = outZtxt + (size_t)N_TXT * DIM;         // [S*N]
    float* outFinal = outAllp + (size_t)S_IMG * N_TXT;       // [N*N]

    char* ws = (char*)d_ws;
    _Float16* zimgH = (_Float16*)ws;                                         // 16 MB
    _Float16* ztxtH = (_Float16*)(ws + (size_t)S_IMG * DIM * 2);             //  4 MB
    _Float16* zselH = (_Float16*)(ws + (size_t)(S_IMG + N_TXT) * DIM * 2);   //  4 MB
    unsigned long long* packed =
        (unsigned long long*)(ws + (size_t)(S_IMG + 2 * N_TXT) * DIM * 2);   // 32 KB
    float* lossAcc = (float*)(packed + N_TXT);

    k_init<<<(N_TXT + 255) / 256, 256, 0, stream>>>(packed, lossAcc);
    k_normalize<<<S_IMG, 128, 0, stream>>>(img, outZimg, zimgH);
    k_normalize<<<N_TXT, 128, 0, stream>>>(txt, outZtxt, ztxtH);

    dim3 g1(N_TXT / 128, S_IMG / 128);
    k_gemm<false><<<g1, 256, 0, stream>>>(zimgH, ztxtH, outAllp, N_TXT, logt, bias, nullptr);

    k_tp<<<(S_IMG + 255) / 256, 256, 0, stream>>>(outAllp, key, packed);
    k_select<<<N_TXT, 128, 0, stream>>>(packed, zimgH, zselH, outSel);

    dim3 g2(N_TXT / 128, N_TXT / 128);
    k_gemm<true><<<g2, 256, 0, stream>>>(zselH, ztxtH, outFinal, N_TXT, logt, bias, lossAcc);

    k_finalize<<<1, 1, 0, stream>>>(lossAcc, outLoss);
}
